// SelfAttention_25074019074498
// MI455X (gfx1250) — matrix operand order repaired and checked by reading, not compiled, not run on hardware
//
#include <hip/hip_runtime.h>

// ---------------- problem constants ----------------
#define EMB   512
#define HEADS 8
#define HSZ   64
#define SEQ   2048
#define BATCH 2
#define ROWS  (BATCH * SEQ)   // 4096

typedef __attribute__((ext_vector_type(16))) __bf16       v16bf;
typedef __attribute__((ext_vector_type(8)))  float        v8f;
typedef __attribute__((ext_vector_type(4)))  unsigned int u32x4;
typedef __attribute__((ext_vector_type(8)))  int          i32x8;
typedef __attribute__((ext_vector_type(4)))  int          i32x4;

union Frag { v16bf v; u32x4 q[2]; unsigned int u[8]; };

#if defined(__has_builtin)
#if __has_builtin(__builtin_amdgcn_tensor_load_to_lds)
#define HAVE_TDM 1
#endif
#endif
#if defined(__has_include)
#if __has_include(<hip/amd_detail/amd_gfx1250_TDM.h>)
#define TDM_6ARG 1
#endif
#endif

__device__ __forceinline__ unsigned int bf16_1(float a) {
  unsigned int x = __float_as_uint(a);
  return (x + 0x7fffu + ((x >> 16) & 1u)) >> 16;   // RNE
}
__device__ __forceinline__ unsigned int pack2bf(float a, float b) {
  return (bf16_1(a) & 0xffffu) | (bf16_1(b) << 16);
}
__device__ __forceinline__ v8f v8f_zero() {
  v8f z;
#pragma unroll
  for (int i = 0; i < 8; ++i) z[i] = 0.0f;
  return z;
}

// ---------------- fp32 -> bf16 conversion ----------------
__global__ __launch_bounds__(256) void cvt_bf16_kernel(const float* __restrict__ src,
                                                       unsigned short* __restrict__ dst,
                                                       int n4) {
  int i = blockIdx.x * blockDim.x + threadIdx.x;
  if (i >= n4) return;
  const float* p = src + (size_t)i * 4;
  unsigned int lo = pack2bf(p[0], p[1]);
  unsigned int hi = pack2bf(p[2], p[3]);
  unsigned int* d = (unsigned int*)dst;
  d[(size_t)i * 2 + 0] = lo;
  d[(size_t)i * 2 + 1] = hi;
}

// ---------------- WMMA 16x64 NT tile, software-pipelined ----------------
__device__ __forceinline__ void ld_frags(const unsigned short* __restrict__ arow,
                                         const unsigned short* __restrict__ B, int ldb,
                                         int col0, int kk, int half, int l16,
                                         Frag& a, Frag b[4]) {
  a.q[0] = *(const u32x4*)(arow + kk + half * 8);
  a.q[1] = *(const u32x4*)(arow + kk + 16 + half * 8);
#pragma unroll
  for (int nb = 0; nb < 4; ++nb) {
    const unsigned short* brow = B + (size_t)(col0 + nb * 16 + l16) * ldb + kk + half * 8;
    b[nb].q[0] = *(const u32x4*)(brow);
    b[nb].q[1] = *(const u32x4*)(brow + 16);
  }
}

// C[16,64] = A[16,K] * B[64,K]^T ; fully unrolled, double-buffered fragments.
template <int K>
__device__ __forceinline__ void gemm16x64(const unsigned short* __restrict__ A, int lda,
                                          const unsigned short* __restrict__ B, int ldb,
                                          int row0, int col0, v8f acc[4]) {
  const int lane = threadIdx.x & 31;
  const int half = lane >> 4;
  const int l16  = lane & 15;
  const unsigned short* arow = A + (size_t)(row0 + l16) * lda;

  Frag a[2];
  Frag b[2][4];
  ld_frags(arow, B, ldb, col0, 0, half, l16, a[0], b[0]);
#pragma unroll
  for (int kk = 0; kk < K; kk += 32) {
    const int cur = (kk / 32) & 1;
    if (kk + 32 < K)
      ld_frags(arow, B, ldb, col0, kk + 32, half, l16, a[cur ^ 1], b[cur ^ 1]);
#pragma unroll
    for (int nb = 0; nb < 4; ++nb)
      acc[nb] = __builtin_amdgcn_wmma_f32_16x16x32_bf16(
          false, a[cur].v, false, b[cur][nb].v, (short)0, acc[nb], false, false);
  }
}

// ---------------- kernel 1: QKV projection + per-head LayerNorm + scale ----------------
__global__ __launch_bounds__(256) void qkv_kernel(
    const unsigned short* __restrict__ xbf,
    const unsigned short* __restrict__ wk, const unsigned short* __restrict__ wq,
    const unsigned short* __restrict__ wv,
    const float* __restrict__ kg, const float* __restrict__ kb,
    const float* __restrict__ qg, const float* __restrict__ qb,
    unsigned short* __restrict__ qbuf,   // [b,h,t,d] bf16
    unsigned short* __restrict__ kbuf,   // [b,h,s,d] bf16
    unsigned short* __restrict__ vtbuf)  // [b,h,d,s] bf16
{
  const int wid   = blockIdx.x * 8 + (threadIdx.x >> 5);   // 0..6143
  const int mat   = wid / (256 * 8);                       // 0=K 1=Q 2=V
  const int rem   = wid % (256 * 8);
  const int mtile = rem >> 3;
  const int htile = rem & 7;
  const int lane = threadIdx.x & 31, half = lane >> 4, l16 = lane & 15;

  const unsigned short* W = (mat == 0) ? wk : ((mat == 1) ? wq : wv);
  v8f acc[4];
#pragma unroll
  for (int i = 0; i < 4; ++i) acc[i] = v8f_zero();

  gemm16x64<EMB>(xbf, EMB, W, EMB, mtile * 16, htile * HSZ, acc);

  const float scale = 0.21022410381342864f;  // 512^(-1/4)
  if (mat != 2) {
    float mu[8], rsig[8];
#pragma unroll
    for (int r = 0; r < 8; ++r) {
      float s = 0.f, ss = 0.f;
#pragma unroll
      for (int nb = 0; nb < 4; ++nb) { float v = acc[nb][r]; s += v; ss += v * v; }
#pragma unroll
      for (int m = 1; m < 16; m <<= 1) {     // stays within half-wave
        s  += __shfl_xor(s, m, 32);
        ss += __shfl_xor(ss, m, 32);
      }
      float mean = s * (1.0f / 64.0f);
      float var  = ss * (1.0f / 64.0f) - mean * mean;
      mu[r]   = mean;
      rsig[r] = __frsqrt_rn(var + 1e-5f);
    }
    const float* g  = (mat == 0) ? kg : qg;
    const float* bb = (mat == 0) ? kb : qb;
#pragma unroll
    for (int nb = 0; nb < 4; ++nb) {
      float gg = g[nb * 16 + l16], bv = bb[nb * 16 + l16];
#pragma unroll
      for (int r = 0; r < 8; ++r)
        acc[nb][r] = ((acc[nb][r] - mu[r]) * rsig[r] * gg + bv) * scale;
    }
  }

  const int trow0 = mtile * 16;
  const int b     = trow0 / SEQ;
  const int tloc0 = trow0 % SEQ;
  if (mat == 2) {
    unsigned short* base = vtbuf + ((size_t)(b * HEADS + htile) * HSZ) * SEQ;
#pragma unroll
    for (int nb = 0; nb < 4; ++nb) {
      int d = nb * 16 + l16;
      u32x4 pk;
      pk.x = pack2bf(acc[nb][0], acc[nb][1]);
      pk.y = pack2bf(acc[nb][2], acc[nb][3]);
      pk.z = pack2bf(acc[nb][4], acc[nb][5]);
      pk.w = pack2bf(acc[nb][6], acc[nb][7]);
      *(u32x4*)(base + (size_t)d * SEQ + tloc0 + half * 8) = pk;
    }
  } else {
    unsigned short* dst =
        ((mat == 0) ? kbuf : qbuf) + ((size_t)(b * HEADS + htile) * SEQ) * HSZ;
#pragma unroll
    for (int nb = 0; nb < 4; ++nb) {
      int d = nb * 16 + l16;
#pragma unroll
      for (int r = 0; r < 8; ++r) {
        int t = tloc0 + r + 8 * half;
        dst[(size_t)t * HSZ + d] = (unsigned short)bf16_1(acc[nb][r]);
      }
    }
  }
}

// ---------------- TDM descriptor issue (2-D tile of 2-byte elements) ----------------
#if defined(HAVE_TDM)
__device__ __forceinline__ void tdm_load_2d(const void* gaddr, unsigned lds_off,
                                            unsigned tile0, unsigned tile1,
                                            unsigned long long stride0_elems,
                                            unsigned tensor_d0, unsigned tensor_d1) {
  unsigned long long ga = (unsigned long long)gaddr;
  u32x4 g0;
  g0.x = 1u;                                            // count=1, user mode
  g0.y = lds_off;                                       // LDS byte address
  g0.z = (unsigned)(ga & 0xffffffffu);                  // global_addr[31:0]
  g0.w = (unsigned)((ga >> 32) & 0x01ffffffu) | (2u << 30);  // addr[56:32] | type=2
  i32x8 g1;
  g1[0] = (int)(1u << 16);                              // data_size=1 (2B), wg_mask=0
  g1[1] = (int)((tensor_d0 & 0xffffu) << 16);           // tensor_dim0[15:0]
  g1[2] = (int)(((tensor_d0 >> 16) & 0xffffu) | ((tensor_d1 & 0xffffu) << 16));
  g1[3] = (int)(((tensor_d1 >> 16) & 0xffffu) | (tile0 << 16));     // tile_dim0
  g1[4] = (int)(tile1 & 0xffffu);                       // tile_dim1 (tile_dim2=0)
  g1[5] = (int)(stride0_elems & 0xffffffffu);           // tensor_dim0_stride[31:0]
  g1[6] = (int)((stride0_elems >> 32) & 0xffffu);       // stride[47:32]
  g1[7] = 0;
  i32x4 z4 = {0, 0, 0, 0};
#if defined(TDM_6ARG)
  i32x8 z8 = {0, 0, 0, 0, 0, 0, 0, 0};
  __builtin_amdgcn_tensor_load_to_lds(g0, g1, z4, z4, z8, 0);
#else
  __builtin_amdgcn_tensor_load_to_lds(g0, g1, z4, z4, 0);
#endif
}
#endif

// ---------------- kernel 2: flash attention ----------------
// One block = one (b,h): 8 waves x 16 queries = 128 queries; K/V^T blocks staged in LDS
// (double-buffered) by the Tensor Data Mover, pipelined one 32-key block ahead.
__global__ __launch_bounds__(256) void attn_kernel(
    const unsigned short* __restrict__ qbuf,
    const unsigned short* __restrict__ kbuf,
    const unsigned short* __restrict__ vtbuf,
    const float* __restrict__ mask,
    unsigned short* __restrict__ obuf)   // [b*T, E] bf16
{
  __shared__ __align__(16) unsigned short sK[2][32 * HSZ];   // [s_local][d]
  __shared__ __align__(16) unsigned short sV[2][HSZ * 32];   // [d][s_local]
  __shared__ __align__(16) unsigned short ldsP[8][16][32];   // per-wave P^T staging

  const int wslot = threadIdx.x >> 5;
  const int bh    = blockIdx.x >> 4;            // 0..15
  const int tblk  = blockIdx.x & 15;
  const int b = bh >> 3, h = bh & 7;
  const int t0 = tblk * 128 + wslot * 16;
  const int lane = threadIdx.x & 31, half = lane >> 4, l16 = lane & 15;

  const unsigned short* qb_ = qbuf + ((size_t)(b * HEADS + h) * SEQ) * HSZ;
  const unsigned short* kb_ = kbuf + ((size_t)(b * HEADS + h) * SEQ) * HSZ;
  const unsigned short* vb_ = vtbuf + ((size_t)(b * HEADS + h) * HSZ) * SEQ;
  const float* mrow = mask + (size_t)b * SEQ;

  // Q as B-fragments of S^T = K * Q^T  (col = t, K-dim = d)
  Frag bq[2];
  {
    const unsigned short* qr = qb_ + (size_t)(t0 + l16) * HSZ + half * 8;
    bq[0].q[0] = *(const u32x4*)(qr);
    bq[0].q[1] = *(const u32x4*)(qr + 16);
    bq[1].q[0] = *(const u32x4*)(qr + 32);
    bq[1].q[1] = *(const u32x4*)(qr + 48);
  }
  const float maskt = mrow[t0 + l16];

  float M = -__builtin_inff(), L = 0.0f;
  v8f ot[4];
#pragma unroll
  for (int i = 0; i < 4; ++i) ot[i] = v8f_zero();

  const int NIT = SEQ / 32;

#if defined(HAVE_TDM)
  // prologue: stage block 0 into buffer 0 (wave 0 drives the TDM)
  if (wslot == 0) {
    tdm_load_2d(kb_, (unsigned)(unsigned long long)&sK[0][0],
                /*tile*/ HSZ, 32, /*stride*/ HSZ, /*tensor*/ HSZ, SEQ);
    tdm_load_2d(vb_, (unsigned)(unsigned long long)&sV[0][0],
                /*tile*/ 32, HSZ, /*stride*/ SEQ, /*tensor*/ SEQ, HSZ);
  }
#else
  {
    int tid = threadIdx.x;
    *(u32x4*)(&sK[0][0] + tid * 8) = *(const u32x4*)(kb_ + tid * 8);
    int row = tid >> 2, off = (tid & 3) * 8;
    *(u32x4*)(&sV[0][0] + row * 32 + off) = *(const u32x4*)(vb_ + (size_t)row * SEQ + off);
  }
#endif

  for (int it = 0; it < NIT; ++it) {
    const int buf = it & 1;
    const int s0  = it * 32;

    __syncthreads();   // everyone finished reading buf^1 (previous iteration)
    if (it + 1 < NIT) {
      const int nb = buf ^ 1;
      const int sn = s0 + 32;
#if defined(HAVE_TDM)
      if (wslot == 0) {
        tdm_load_2d(kb_ + (size_t)sn * HSZ, (unsigned)(unsigned long long)&sK[nb][0],
                    HSZ, 32, HSZ, HSZ, SEQ);
        tdm_load_2d(vb_ + sn, (unsigned)(unsigned long long)&sV[nb][0],
                    32, HSZ, SEQ, SEQ, HSZ);
      }
#else
      int tid = threadIdx.x;
      *(u32x4*)(&sK[nb][0] + tid * 8) = *(const u32x4*)(kb_ + (size_t)sn * HSZ + tid * 8);
      int row = tid >> 2, off = (tid & 3) * 8;
      *(u32x4*)(&sV[nb][0] + row * 32 + off) =
          *(const u32x4*)(vb_ + (size_t)row * SEQ + sn + off);
#endif
    }
#if defined(HAVE_TDM)
    if (wslot == 0) {
      if (it + 1 < NIT) __builtin_amdgcn_s_wait_tensorcnt(2);   // current pair done
      else              __builtin_amdgcn_s_wait_tensorcnt(0);
    }
#endif
    __syncthreads();   // staged data for `buf` visible to all waves

    const unsigned short* kS = &sK[buf][0];
    const unsigned short* vS = &sV[buf][0];

    // S^T block: 32 keys x 16 queries; K A-fragments from LDS
    v8f sacc[2];
#pragma unroll
    for (int sub = 0; sub < 2; ++sub) {
      const unsigned short* kr = kS + (size_t)(sub * 16 + l16) * HSZ;
      Frag a0, a1;
      a0.q[0] = *(const u32x4*)(kr + half * 8);
      a0.q[1] = *(const u32x4*)(kr + 16 + half * 8);
      a1.q[0] = *(const u32x4*)(kr + 32 + half * 8);
      a1.q[1] = *(const u32x4*)(kr + 48 + half * 8);
      v8f s = __builtin_amdgcn_wmma_f32_16x16x32_bf16(
          false, a0.v, false, bq[0].v, (short)0, v8f_zero(), false, false);
      sacc[sub] = __builtin_amdgcn_wmma_f32_16x16x32_bf16(
          false, a1.v, false, bq[1].v, (short)0, s, false, false);
    }
    // masking + online softmax (row t = l16; s spread over r and the half pair)
    float blkmax = -__builtin_inff();
#pragma unroll
    for (int sub = 0; sub < 2; ++sub)
#pragma unroll
      for (int r = 0; r < 8; ++r) {
        int s = s0 + sub * 16 + r + 8 * half;
        float v = sacc[sub][r];
        if (!(mrow[s] > 0.0f && maskt > 0.0f)) v = -__builtin_inff();
        sacc[sub][r] = v;
        blkmax = fmaxf(blkmax, v);
      }
    blkmax = fmaxf(blkmax, __shfl_xor(blkmax, 16, 32));
    float newM  = fmaxf(M, blkmax);
    float alpha = (newM == -__builtin_inff()) ? 1.0f : __expf(M - newM);
    float lsum = 0.0f;
#pragma unroll
    for (int sub = 0; sub < 2; ++sub)
#pragma unroll
      for (int r = 0; r < 8; ++r) {
        float v = sacc[sub][r];
        float e = (v == -__builtin_inff()) ? 0.0f : __expf(v - newM);
        sacc[sub][r] = e;
        lsum += e;
      }
    lsum += __shfl_xor(lsum, 16, 32);
    L = L * alpha + lsum;
    M = newM;
#pragma unroll
    for (int d = 0; d < 4; ++d)
#pragma unroll
      for (int r = 0; r < 8; ++r) ot[d][r] *= alpha;

    // stage P^T in (wave-private) LDS as [t][s_local] bf16, reload as B-fragments
#pragma unroll
    for (int sub = 0; sub < 2; ++sub) {
      u32x4 pk;
      pk.x = pack2bf(sacc[sub][0], sacc[sub][1]);
      pk.y = pack2bf(sacc[sub][2], sacc[sub][3]);
      pk.z = pack2bf(sacc[sub][4], sacc[sub][5]);
      pk.w = pack2bf(sacc[sub][6], sacc[sub][7]);
      *(u32x4*)&ldsP[wslot][l16][sub * 16 + half * 8] = pk;
    }
    asm volatile("s_wait_dscnt 0x0" ::: "memory");   // wave-private LDS RAW
    Frag bp;
    bp.q[0] = *(const u32x4*)&ldsP[wslot][l16][half * 8];
    bp.q[1] = *(const u32x4*)&ldsP[wslot][l16][half * 8 + 16];

    // O^T += V^T(16d x 32s) * P^T(32s x 16t); V fragments from LDS
#pragma unroll
    for (int dt = 0; dt < 4; ++dt) {
      const unsigned short* vr = vS + (size_t)(dt * 16 + l16) * 32;
      Frag av;
      av.q[0] = *(const u32x4*)(vr + half * 8);
      av.q[1] = *(const u32x4*)(vr + 16 + half * 8);
      ot[dt] = __builtin_amdgcn_wmma_f32_16x16x32_bf16(
          false, av.v, false, bp.v, (short)0, ot[dt], false, false);
    }
  }

  const float inv = (L > 0.0f) ? (1.0f / L) : 0.0f;
  unsigned short* orow = obuf + ((size_t)(b * SEQ + t0 + l16)) * EMB + h * HSZ;
#pragma unroll
  for (int dt = 0; dt < 4; ++dt) {
    u32x4 pk;
    pk.x = pack2bf(ot[dt][0] * inv, ot[dt][1] * inv);
    pk.y = pack2bf(ot[dt][2] * inv, ot[dt][3] * inv);
    pk.z = pack2bf(ot[dt][4] * inv, ot[dt][5] * inv);
    pk.w = pack2bf(ot[dt][6] * inv, ot[dt][7] * inv);
    *(u32x4*)(orow + dt * 16 + half * 8) = pk;
  }
}

// ---------------- kernel 3: output projection + bias (fp32 out) ----------------
__global__ __launch_bounds__(256) void proj_kernel(
    const unsigned short* __restrict__ obuf,
    const unsigned short* __restrict__ wu,
    const float* __restrict__ bu,
    float* __restrict__ out)
{
  const int wid   = blockIdx.x * 8 + (threadIdx.x >> 5);   // 0..2047
  const int mtile = wid >> 3;
  const int ntile = wid & 7;
  const int lane = threadIdx.x & 31, half = lane >> 4, l16 = lane & 15;

  v8f acc[4];
#pragma unroll
  for (int i = 0; i < 4; ++i) acc[i] = v8f_zero();
  gemm16x64<EMB>(obuf, EMB, wu, EMB, mtile * 16, ntile * 64, acc);

  const int row0 = mtile * 16;
#pragma unroll
  for (int nb = 0; nb < 4; ++nb) {
    int e = ntile * 64 + nb * 16 + l16;
    float bv = bu[e];
#pragma unroll
    for (int r = 0; r < 8; ++r)
      out[(size_t)(row0 + r + 8 * half) * EMB + e] = acc[nb][r] + bv;
  }
}

// ---------------- host launcher ----------------
extern "C" void kernel_launch(void* const* d_in, const int* in_sizes, int n_in,
                              void* d_out, int out_size, void* d_ws, size_t ws_size,
                              hipStream_t stream) {
  const float* x    = (const float*)d_in[0];
  const float* mask = (const float*)d_in[1];
  const float* Wk   = (const float*)d_in[2];
  const float* Wq   = (const float*)d_in[3];
  const float* Wv   = (const float*)d_in[4];
  const float* Wu   = (const float*)d_in[5];
  const float* bu   = (const float*)d_in[6];
  const float* kg   = (const float*)d_in[7];
  const float* kb   = (const float*)d_in[8];
  const float* qg   = (const float*)d_in[9];
  const float* qb   = (const float*)d_in[10];
  float* out = (float*)d_out;

  char* ws = (char*)d_ws;
  unsigned short* xbf   = (unsigned short*)ws; ws += (size_t)ROWS * EMB * 2;
  unsigned short* wkbf  = (unsigned short*)ws; ws += (size_t)EMB * EMB * 2;
  unsigned short* wqbf  = (unsigned short*)ws; ws += (size_t)EMB * EMB * 2;
  unsigned short* wvbf  = (unsigned short*)ws; ws += (size_t)EMB * EMB * 2;
  unsigned short* wubf  = (unsigned short*)ws; ws += (size_t)EMB * EMB * 2;
  unsigned short* qbuf  = (unsigned short*)ws; ws += (size_t)ROWS * EMB * 2;
  unsigned short* kbuf  = (unsigned short*)ws; ws += (size_t)ROWS * EMB * 2;
  unsigned short* vtbuf = (unsigned short*)ws; ws += (size_t)ROWS * EMB * 2;
  unsigned short* obuf  = (unsigned short*)ws; ws += (size_t)ROWS * EMB * 2;

  {
    int n4 = (ROWS * EMB) / 4;
    cvt_bf16_kernel<<<n4 / 256, 256, 0, stream>>>(x, xbf, n4);
    int w4 = (EMB * EMB) / 4;
    cvt_bf16_kernel<<<w4 / 256, 256, 0, stream>>>(Wk, wkbf, w4);
    cvt_bf16_kernel<<<w4 / 256, 256, 0, stream>>>(Wq, wqbf, w4);
    cvt_bf16_kernel<<<w4 / 256, 256, 0, stream>>>(Wv, wvbf, w4);
    cvt_bf16_kernel<<<w4 / 256, 256, 0, stream>>>(Wu, wubf, w4);
  }
  qkv_kernel<<<768, 256, 0, stream>>>(xbf, wkbf, wqbf, wvbf, kg, kb, qg, qb,
                                      qbuf, kbuf, vtbuf);
  attn_kernel<<<256, 256, 0, stream>>>(qbuf, kbuf, vtbuf, mask, obuf);
  proj_kernel<<<256, 256, 0, stream>>>(obuf, wubf, bu, out);
}
